// DiagonalMicroAttention_87497073754696
// MI455X (gfx1250) — matrix operand order repaired and checked by reading, not compiled, not run on hardware
//
#include <hip/hip_runtime.h>
#include <hip/hip_bf16.h>
#include <math.h>

// Problem constants (from reference)
#define BB 2
#define CC 128
#define HH 56
#define WW 56
#define NN (HH * WW)      // 3136 spatial positions
#define NHEADS 4
#define DH (CC / NHEADS)  // 32
#define W2 (WW / 2)       // 28
#define P2 (HH * W2)      // 1568 positions per batch in gating branch

typedef __attribute__((ext_vector_type(16))) __bf16 v16bf;
typedef __attribute__((ext_vector_type(8)))  float  v8f;
typedef __attribute__((ext_vector_type(4)))  unsigned int v4u;

union Frag { v16bf v; v4u u[2]; };

__device__ __forceinline__ unsigned short f2bf(float f) {
  unsigned int u = __float_as_uint(f);
  u += 0x7FFFu + ((u >> 16) & 1u);   // round-to-nearest-even
  return (unsigned short)(u >> 16);
}

// ---------------------------------------------------------------------------
// Conversions
// ---------------------------------------------------------------------------
// x: (B,C,H,W) f32  ->  Xbf: [b][n][c] bf16 (GEMM A operand, row-major N x C)
__global__ void cvt_x_kernel(const float* __restrict__ x, unsigned short* __restrict__ Xbf) {
  int idx = blockIdx.x * blockDim.x + threadIdx.x;
  if (idx >= BB * NN * CC) return;
  int c = idx % CC;
  int n = (idx / CC) % NN;
  int b = idx / (CC * NN);
  Xbf[idx] = f2bf(x[((size_t)b * CC + c) * NN + n]);
}

__global__ void cvt_w_kernel(const float* __restrict__ W, unsigned short* __restrict__ Wbf, int count) {
  int idx = blockIdx.x * blockDim.x + threadIdx.x;
  if (idx < count) Wbf[idx] = f2bf(W[idx]);
}

// ---------------------------------------------------------------------------
// WMMA 16x16 tile GEMM core: D(16x16,f32) = A(16x128,bf16) * Wt(16 rows x 128)^T
// A row-major [M x 128], Wt row-major [O x 128]; D[m][o] = sum_k A[m,k]*Wt[o,k]
// Fragment layouts per CDNA5 ISA 7.12.2 (16-bit A 16x32, 16-bit B 32x16).
// ---------------------------------------------------------------------------
__device__ __forceinline__ v8f gemm_tile_bf16(const unsigned short* __restrict__ A,
                                              const unsigned short* __restrict__ Wt,
                                              int tileM, int tileO) {
  const int lane = threadIdx.x & 31;
  const int m16  = lane & 15;
  const int g    = lane >> 4;
  v8f acc = {0.f, 0.f, 0.f, 0.f, 0.f, 0.f, 0.f, 0.f};
  const unsigned short* arow = A  + (size_t)(tileM * 16 + m16) * CC;
  const unsigned short* brow = Wt + (size_t)(tileO * 16 + m16) * CC;
#pragma unroll
  for (int kb = 0; kb < CC; kb += 32) {
    Frag a, b;
    // A: lane(m16,g): elems 0-7 <- K = kb+g*8..+7 ; elems 8-15 <- K = kb+16+g*8..+7
    a.u[0] = *reinterpret_cast<const v4u*>(arow + kb + g * 8);
    a.u[1] = *reinterpret_cast<const v4u*>(arow + kb + 16 + g * 8);
    // B: lane(n=m16,g): elems 0-7 <- K = kb+g*8..+7 ; elems 8-15 <- K = kb+16+g*8..+7
    b.u[0] = *reinterpret_cast<const v4u*>(brow + kb + g * 8);
    b.u[1] = *reinterpret_cast<const v4u*>(brow + kb + 16 + g * 8);
    acc = __builtin_amdgcn_wmma_f32_16x16x32_bf16(false, a.v, false, b.v,
                                                  (short)0, acc, false, false);
  }
  return acc;
}

// QKV projection: grid(196, B, 3), 256 thr = 8 waves -> all 128 out channels.
__global__ void gemm_qkv_kernel(const unsigned short* __restrict__ Xbf,
                                const unsigned short* __restrict__ Wqb,
                                const unsigned short* __restrict__ Wkb,
                                const unsigned short* __restrict__ Wvb,
                                float* __restrict__ Q, float* __restrict__ K,
                                float* __restrict__ V) {
  const int tileM = blockIdx.x;
  const int b     = blockIdx.y;
  const int which = blockIdx.z;
  const int tileO = threadIdx.x >> 5;  // wave id 0..7
  const unsigned short* Wt = (which == 0) ? Wqb : (which == 1) ? Wkb : Wvb;
  float* Out               = (which == 0) ? Q   : (which == 1) ? K   : V;
  const unsigned short* A = Xbf + (size_t)b * NN * CC;
  v8f acc = gemm_tile_bf16(A, Wt, tileM, tileO);
  const int lane = threadIdx.x & 31;
  const int col  = lane & 15;
  const int half = lane >> 4;
  float* ob = Out + (size_t)b * NN * CC;
#pragma unroll
  for (int r = 0; r < 8; ++r) {
    int mm = tileM * 16 + r + half * 8;           // D: vgpr r, lanes0-15 M=r, 16-31 M=8+r
    ob[(size_t)mm * CC + tileO * 16 + col] = acc[r];
  }
}

// Output projection + bias, scatter to (B,C,H,W): grid(196, B)
__global__ void gemm_out_kernel(const unsigned short* __restrict__ AObf,
                                const unsigned short* __restrict__ Wob,
                                const float* __restrict__ bo,
                                float* __restrict__ out) {
  const int tileM = blockIdx.x;
  const int b     = blockIdx.y;
  const int tileO = threadIdx.x >> 5;
  const unsigned short* A = AObf + (size_t)b * NN * CC;
  v8f acc = gemm_tile_bf16(A, Wob, tileM, tileO);
  const int lane = threadIdx.x & 31;
  const int col  = lane & 15;
  const int half = lane >> 4;
  const int o    = tileO * 16 + col;
  const float bias = bo[o];
  float* ob = out + (size_t)b * CC * NN + (size_t)o * NN;
#pragma unroll
  for (int r = 0; r < 8; ++r) {
    int mm = tileM * 16 + r + half * 8;
    ob[mm] = acc[r] + bias;
  }
}

// ---------------------------------------------------------------------------
// Gating (asym) branch
// ---------------------------------------------------------------------------
// hidden[b][o][p] = gelu( ba1[o] + sum_c Wa1[o,c]*cat[b,c,p] ), p = h*28+w2
__global__ void asym_hidden_kernel(const float* __restrict__ x,
                                   const float* __restrict__ Wa1,
                                   const float* __restrict__ ba1,
                                   float* __restrict__ hid) {
  int idx = blockIdx.x * blockDim.x + threadIdx.x;
  if (idx >= BB * CC * P2) return;
  int p = idx % P2;
  int o = (idx / P2) % CC;
  int b = idx / (P2 * CC);
  int h = p / W2, w2 = p % W2;
  const float* xb = x + (size_t)b * CC * NN;
  const float* wr = Wa1 + (size_t)o * (2 * CC);
  float acc = ba1[o];
#pragma unroll 4
  for (int c = 0; c < CC; ++c)          // left half
    acc += wr[c] * xb[(size_t)c * NN + h * WW + w2];
#pragma unroll 4
  for (int c = 0; c < CC; ++c)          // right half, W-flipped
    acc += wr[CC + c] * xb[(size_t)c * NN + h * WW + (WW - 1 - w2)];
  // exact GELU: 0.5*x*(1+erf(x/sqrt(2)))
  hid[idx] = 0.5f * acc * (1.f + erff(acc * 0.70710678118654752f));
}

// asymH[b][p] = sigmoid( ba2 + sum_o Wa2[o]*hid[b][o][p] )
__global__ void asym_reduce_kernel(const float* __restrict__ hid,
                                   const float* __restrict__ Wa2,
                                   const float* __restrict__ ba2,
                                   float* __restrict__ asymH) {
  int idx = blockIdx.x * blockDim.x + threadIdx.x;
  if (idx >= BB * P2) return;
  int p = idx % P2;
  int b = idx / P2;
  float acc = ba2[0];
#pragma unroll 4
  for (int o = 0; o < CC; ++o)
    acc += Wa2[o] * hid[((size_t)(b * CC + o)) * P2 + p];
  asymH[idx] = 1.f / (1.f + expf(-acc));
}

// half-pixel linear resize 28 -> 56 along W (jax.image.resize 'linear')
__global__ void asym_resize_kernel(const float* __restrict__ asymH,
                                   float* __restrict__ asymF) {
  int idx = blockIdx.x * blockDim.x + threadIdx.x;
  if (idx >= BB * NN) return;
  int w = idx % WW;
  int h = (idx / WW) % HH;
  int b = idx / NN;
  float src = 0.5f * (float)w - 0.25f;
  src = fminf(fmaxf(src, 0.f), (float)(W2 - 1));
  int i0 = (int)floorf(src);
  int i1 = min(i0 + 1, W2 - 1);
  float t = src - (float)i0;
  const float* r = asymH + (size_t)b * P2 + h * W2;
  asymF[idx] = (1.f - t) * r[i0] + t * r[i1];
}

// ---------------------------------------------------------------------------
// Local 3x3-window softmax attention, fused with asym gate. One thread per
// (b, head, query position). All data stays in VGPRs (q[32], out[32], s[9]).
// Equivalent to the reference's masked dense softmax: masked entries get
// exp(-1e9 - m) ~ 0 and every position has >= 4 valid neighbors.
// ---------------------------------------------------------------------------
__global__ void attn_kernel(const float* __restrict__ Qf, const float* __restrict__ Kf,
                            const float* __restrict__ Vf, const float* __restrict__ asymF,
                            unsigned short* __restrict__ AObf) {
  int idx = blockIdx.x * blockDim.x + threadIdx.x;
  if (idx >= BB * NHEADS * NN) return;
  int n    = idx % NN;
  int head = (idx / NN) % NHEADS;
  int b    = idx / (NN * NHEADS);
  int ii = n / WW, jj = n % WW;
  const float scale = 0.17677669529663687f;  // dh^-0.5, dh=32

  const float* qp = Qf + ((size_t)b * NN + n) * CC + head * DH;
  float q[DH];
#pragma unroll
  for (int d = 0; d < DH; ++d) q[d] = qp[d];

  float s[9];
  int nb[9];
  int cnt = 0;
  float mx = -1e30f;
  for (int di = -1; di <= 1; ++di)
    for (int dj = -1; dj <= 1; ++dj) {
      int i2 = ii + di, j2 = jj + dj;
      if (i2 < 0 || i2 >= HH || j2 < 0 || j2 >= WW) continue;
      int j = i2 * WW + j2;
      const float* kp = Kf + ((size_t)b * NN + j) * CC + head * DH;
      float acc = 0.f;
#pragma unroll
      for (int d = 0; d < DH; ++d) acc += q[d] * kp[d];
      acc *= scale;
      s[cnt] = acc;
      nb[cnt] = j;
      mx = fmaxf(mx, acc);
      ++cnt;
    }
  float sum = 0.f;
  for (int t = 0; t < cnt; ++t) { s[t] = expf(s[t] - mx); sum += s[t]; }
  float inv = 1.f / sum;

  float out[DH];
#pragma unroll
  for (int d = 0; d < DH; ++d) out[d] = 0.f;
  for (int t = 0; t < cnt; ++t) {
    const float* vp = Vf + ((size_t)b * NN + nb[t]) * CC + head * DH;
    float p = s[t] * inv;
#pragma unroll
    for (int d = 0; d < DH; ++d) out[d] += p * vp[d];
  }

  float gate = 1.f + 0.5f * asymF[b * NN + n];
  unsigned short* op = AObf + ((size_t)b * NN + n) * CC + head * DH;
#pragma unroll
  for (int d = 0; d < DH; ++d) op[d] = f2bf(out[d] * gate);
}

// ---------------------------------------------------------------------------
// Launch
// ---------------------------------------------------------------------------
extern "C" void kernel_launch(void* const* d_in, const int* in_sizes, int n_in,
                              void* d_out, int out_size, void* d_ws, size_t ws_size,
                              hipStream_t stream) {
  const float* x   = (const float*)d_in[0];
  const float* Wq  = (const float*)d_in[1];
  const float* Wk  = (const float*)d_in[2];
  const float* Wv  = (const float*)d_in[3];
  const float* Wo  = (const float*)d_in[4];
  const float* bo  = (const float*)d_in[5];
  const float* Wa1 = (const float*)d_in[6];
  const float* ba1 = (const float*)d_in[7];
  const float* Wa2 = (const float*)d_in[8];
  const float* ba2 = (const float*)d_in[9];
  float* out = (float*)d_out;

  // Workspace layout (bytes, all 256B aligned)
  char* ws = (char*)d_ws;
  unsigned short* Xbf  = (unsigned short*)(ws + 0);          // 1,605,632 B
  unsigned short* Wqb  = (unsigned short*)(ws + 1605632);    //    32,768 B
  unsigned short* Wkb  = (unsigned short*)(ws + 1638400);
  unsigned short* Wvb  = (unsigned short*)(ws + 1671168);
  unsigned short* Wob  = (unsigned short*)(ws + 1703936);
  float* Qf            = (float*)(ws + 1736704);             // 3,211,264 B
  float* Kf            = (float*)(ws + 4947968);
  float* Vf            = (float*)(ws + 8159232);
  float* hid           = (float*)(ws + 11370496);            // 1,605,632 B
  float* asymH         = (float*)(ws + 12976128);            //    12,544 B
  float* asymF         = (float*)(ws + 12988672);            //    25,088 B
  unsigned short* AObf = (unsigned short*)(ws + 13013760);   // 1,605,632 B

  const int T = 256;

  // 1) bf16 conversions
  cvt_x_kernel<<<(BB * NN * CC + T - 1) / T, T, 0, stream>>>(x, Xbf);
  cvt_w_kernel<<<(CC * CC + T - 1) / T, T, 0, stream>>>(Wq, Wqb, CC * CC);
  cvt_w_kernel<<<(CC * CC + T - 1) / T, T, 0, stream>>>(Wk, Wkb, CC * CC);
  cvt_w_kernel<<<(CC * CC + T - 1) / T, T, 0, stream>>>(Wv, Wvb, CC * CC);
  cvt_w_kernel<<<(CC * CC + T - 1) / T, T, 0, stream>>>(Wo, Wob, CC * CC);

  // 2) gating branch
  asym_hidden_kernel<<<(BB * CC * P2 + T - 1) / T, T, 0, stream>>>(x, Wa1, ba1, hid);
  asym_reduce_kernel<<<(BB * P2 + T - 1) / T, T, 0, stream>>>(hid, Wa2, ba2, asymH);
  asym_resize_kernel<<<(BB * NN + T - 1) / T, T, 0, stream>>>(asymH, asymF);

  // 3) QKV projections via WMMA (grid: 196 M-tiles x B x {q,k,v})
  gemm_qkv_kernel<<<dim3(NN / 16, BB, 3), T, 0, stream>>>(Xbf, Wqb, Wkb, Wvb, Qf, Kf, Vf);

  // 4) local 3x3 attention + gate -> bf16 activations
  attn_kernel<<<(BB * NHEADS * NN + T - 1) / T, T, 0, stream>>>(Qf, Kf, Vf, asymF, AObf);

  // 5) output projection via WMMA, scatter to (B,C,H,W) + bias
  gemm_out_kernel<<<dim3(NN / 16, BB, 1), T, 0, stream>>>(AObf, Wob, bo, out);
}
